// RGCN_47064251630182
// MI455X (gfx1250) — matrix operand order repaired and checked by reading, not compiled, not run on hardware
//
#include <hip/hip_runtime.h>
#include <hip/hip_bf16.h>
#include <stdint.h>

#define DIM 128   // DIN = DH = DOUT = 128

// ---- CDNA5 WMMA vector types (wave32) ----
typedef __attribute__((ext_vector_type(16))) __bf16 bf16x16;
typedef __attribute__((ext_vector_type(8)))  __bf16 bf16x8;
typedef __attribute__((ext_vector_type(8)))  float  f32x8;

// ---------------- per-(relation,node) in-degree, layout [R][N] ----------------
__global__ void k_count(const int* __restrict__ dst, const int* __restrict__ et,
                        int E, int N, int* __restrict__ cnt) {
    int e = blockIdx.x * blockDim.x + threadIdx.x;
    if (e < E) atomicAdd(&cnt[(size_t)et[e] * N + dst[e]], 1);
}

__global__ void k_inv(int n, const int* __restrict__ cnt, float* __restrict__ inv) {
    int i = blockIdx.x * blockDim.x + threadIdx.x;
    if (i < n) {
        int c = cnt[i];
        inv[i] = 1.0f / (float)(c < 1 ? 1 : c);   // 1/clip(deg,1)
    }
}

// ---------------- h0 = emb[x] ----------------
__global__ void k_gather(const int* __restrict__ x, const float* __restrict__ emb,
                         float* __restrict__ h, long total) {
    long i = (long)blockIdx.x * blockDim.x + threadIdx.x;
    if (i < total) {
        long node = i >> 7, c = i & 127;
        h[i] = emb[(size_t)x[node] * DIM + c];
    }
}

// ---------------- weights -> bf16, transposed to [slot][n][k] ----------------
// slot 0..R-1 = w[r]; slot R = root.
__global__ void k_wconv(const float* __restrict__ w, const float* __restrict__ root,
                        __bf16* __restrict__ wT, int R) {
    long i = (long)blockIdx.x * blockDim.x + threadIdx.x;
    long total = (long)(R + 1) * DIM * DIM;
    if (i >= total) return;
    int slot = (int)(i / (DIM * DIM));
    int rem  = (int)(i % (DIM * DIM));
    int n = rem / DIM, k = rem % DIM;
    float v = (slot < R) ? w[(size_t)slot * DIM * DIM + (size_t)k * DIM + n]
                         : root[(size_t)k * DIM + n];
    wT[i] = (__bf16)v;    // native cvt
}

__global__ void k_relu(const float* __restrict__ a, float* __restrict__ b, long n) {
    long i = (long)blockIdx.x * blockDim.x + threadIdx.x;
    if (i < n) { float v = a[i]; b[i] = v > 0.0f ? v : 0.0f; }
}

// ---------------- relation-filtered segment sum: agg[dst] += h[src] ----------------
// 128 threads = one feature row; 32 edges per block; edge-type read is a
// uniform (broadcast) load, atomic adds are fully coalesced per edge.
__global__ void k_agg(const int* __restrict__ src, const int* __restrict__ dst,
                      const int* __restrict__ et, int E, int rel,
                      const float* __restrict__ h, float* __restrict__ agg) {
    const int t  = threadIdx.x;              // 0..127 feature channel
    const int e0 = blockIdx.x * 32;
    const int e1 = (e0 + 32 < E) ? e0 + 32 : E;
    for (int e = e0; e < e1; ++e) {
        if (et[e] == rel) {
            int s = src[e], d = dst[e];
            atomicAdd(&agg[(size_t)d * DIM + t], h[(size_t)s * DIM + t]);
        }
    }
}

// ---------------- WMMA GEMM: 16-row strip x 128 cols per block (8 waves) ----
// A (f32 [M][128]) is staged once per block into LDS as bf16 by all 256
// threads, then each wave reads its fragments with ds_load_b128.
// mode 0:  O[m][n] = (A @ Wt)[m][n] + bias[n]
// mode 1:  O[m][n] += inv[rel*M+m] * (A @ Wt)[m][n]
//          (per-row mean scale commutes with the GEMM -> folded into the 8
//           C elements; inv is [R][M] so the 8 scales are contiguous)
// Fragment layouts used here (wave32; A/B k order is contiguous, not the ISA table order):
//   A 16x32 bf16 : lane m = l&15, hi = l>>4; half i -> K = hi*16+i
//   B 32x16 bf16 : lane n = l&15; half t -> K = hi*16+t   (Wt stored [n][k] => contiguous 32B)
//   C/D 16x16 f32: vgpr j -> M = hi*8+j, N = l&15
__global__ void __launch_bounds__(256)
k_gemm16(const float*  __restrict__ A,        // [M][128] f32
         const float*  __restrict__ invdeg,   // [R][M] (mode 1)
         int rel,
         const __bf16* __restrict__ Wt,       // [128][128] bf16, [n][k]
         const float*  __restrict__ bias,     // (mode 0)
         float* __restrict__ O,
         int M, int mode) {
    __shared__ __bf16 As[16 * DIM];           // 4 KB

    const int tid  = threadIdx.x;
    const int lane = tid & 31;
    const int wave = tid >> 5;                // 0..7 -> 16-col tile
    const int hi   = lane >> 4;
    const int m16  = lane & 15;
    const int row0 = blockIdx.x * 16;
    const int col  = wave * 16 + m16;
    const bool full = (row0 + 16 <= M);       // uniform: strip fully in range

    // ---- cooperative A stage: 16x128 f32 -> bf16 in LDS (8 elems/thread) ----
    {
        const float* ap = A + (size_t)row0 * DIM + tid;   // + j*256 below
        if (full) {
            #pragma unroll
            for (int j = 0; j < 8; ++j)
                As[j * 256 + tid] = (__bf16)ap[j * 256];
        } else {
            #pragma unroll
            for (int j = 0; j < 8; ++j) {
                int i = j * 256 + tid;
                float v = (row0 + (i >> 7) < M) ? ap[j * 256] : 0.0f;
                As[i] = (__bf16)v;
            }
        }
    }
    __syncthreads();

    // ---- 16x16x128 via 4 x wmma 16x16x32 ----
    f32x8 c = {};
    #pragma unroll
    for (int kt = 0; kt < DIM / 32; ++kt) {
        bf16x8 a0 = *(const bf16x8*)(As + m16 * DIM + kt * 32 + hi * 16);
        bf16x8 a1 = *(const bf16x8*)(As + m16 * DIM + kt * 32 + hi * 16 + 8);
        bf16x16 a;
        #pragma unroll
        for (int i = 0; i < 8; ++i) { a[i] = a0[i]; a[8 + i] = a1[i]; }
        bf16x16 b = *(const bf16x16*)(Wt + (size_t)col * DIM + kt * 32 + hi * 16);
        c = __builtin_amdgcn_wmma_f32_16x16x32_bf16(false, a, false, b,
                                                    (short)0, c, false, false);
    }

    // ---- epilogue: base pointer + immediate offsets; no per-element guards
    //      on the (always-taken for M%16==0) full path ----
    const int mr0 = row0 + hi * 8;            // first of this half-wave's 8 rows
    float* op = O + (size_t)mr0 * DIM + col;  // row stride DIM (imm offsets)
    if (mode == 0) {
        const float badd = bias[col];
        if (full) {
            #pragma unroll
            for (int j = 0; j < 8; ++j) op[j * DIM] = c[j] + badd;
        } else {
            #pragma unroll
            for (int j = 0; j < 8; ++j)
                if (mr0 + j < M) op[j * DIM] = c[j] + badd;
        }
    } else {
        const float* ip = invdeg + (size_t)rel * M + mr0;  // 8 contiguous scales
        if (full) {
            #pragma unroll
            for (int j = 0; j < 8; ++j)
                op[j * DIM] = op[j * DIM] + ip[j] * c[j];
        } else {
            #pragma unroll
            for (int j = 0; j < 8; ++j)
                if (mr0 + j < M) op[j * DIM] = op[j * DIM] + ip[j] * c[j];
        }
    }
}

extern "C" void kernel_launch(void* const* d_in, const int* in_sizes, int n_in,
                              void* d_out, int out_size, void* d_ws, size_t ws_size,
                              hipStream_t stream) {
    const int N = in_sizes[0];
    const int E = in_sizes[2];
    const int R = in_sizes[4] / (DIM * DIM);   // 20

    const int*   x    = (const int*)d_in[0];
    const int*   src  = (const int*)d_in[1];   // edge[0]
    const int*   dstp = src + E;               // edge[1]
    const int*   et   = (const int*)d_in[2];
    const float* emb  = (const float*)d_in[3];
    const float* w1   = (const float*)d_in[4];
    const float* r1   = (const float*)d_in[5];
    const float* b1   = (const float*)d_in[6];
    const float* w2   = (const float*)d_in[7];
    const float* r2   = (const float*)d_in[8];
    const float* b2   = (const float*)d_in[9];
    float* O = (float*)d_out;

    // ---- carve workspace (~113 MB) ----
    uintptr_t p = (uintptr_t)d_ws;
    auto carve = [&](size_t bytes) -> void* {
        p = (p + 255) & ~(uintptr_t)255;
        void* q = (void*)p; p += bytes; return q;
    };
    int*    cnt  = (int*)carve((size_t)N * R * 4);          // [R][N], reused as inv
    float*  inv  = (float*)cnt;
    float*  hbuf = (float*)carve((size_t)N * DIM * 4);      // current layer input (f32)
    float*  agg  = (float*)carve((size_t)N * DIM * 4);      // per-relation segment sum
    __bf16* wT1  = (__bf16*)carve((size_t)(R + 1) * DIM * DIM * 2);
    __bf16* wT2  = (__bf16*)carve((size_t)(R + 1) * DIM * DIM * 2);
    (void)ws_size; (void)n_in; (void)out_size;

    const int  TPB   = 256;
    const long NH    = (long)N * DIM;
    const int  gElem = (int)((NH + TPB - 1) / TPB);
    const int  gW    = (int)(((long)(R + 1) * DIM * DIM + TPB - 1) / TPB);
    const int  gGemm = (N + 15) / 16;

    // ---- 1/deg per (relation, node) ----
    hipMemsetAsync(cnt, 0, (size_t)N * R * 4, stream);
    k_count<<<(E + TPB - 1) / TPB, TPB, 0, stream>>>(dstp, et, E, N, cnt);
    k_inv<<<(N * R + TPB - 1) / TPB, TPB, 0, stream>>>(N * R, cnt, inv);

    // ---- h0 = emb[x] ----
    k_gather<<<gElem, TPB, 0, stream>>>(x, emb, hbuf, NH);

    // ---- weights to bf16 transposed ----
    k_wconv<<<gW, TPB, 0, stream>>>(w1, r1, wT1, R);
    k_wconv<<<gW, TPB, 0, stream>>>(w2, r2, wT2, R);

    auto layer = [&](const float* hin, const __bf16* wT, const float* bias, float* out) {
        // root transform initializes out (also un-poisons d_out fully)
        k_gemm16<<<gGemm, TPB, 0, stream>>>(hin, nullptr, 0,
                                            wT + (size_t)R * DIM * DIM, bias, out, N, 0);
        for (int r = 0; r < R; ++r) {
            hipMemsetAsync(agg, 0, (size_t)N * DIM * 4, stream);
            k_agg<<<(E + 31) / 32, DIM, 0, stream>>>(src, dstp, et, E, r, hin, agg);
            k_gemm16<<<gGemm, TPB, 0, stream>>>(agg, inv, r,
                                                wT + (size_t)r * DIM * DIM, nullptr, out, N, 1);
        }
    };

    layer(hbuf, wT1, b1, O);                 // h1 -> d_out (scratch)
    k_relu<<<gElem, TPB, 0, stream>>>(O, hbuf, NH);
    layer(hbuf, wT2, b2, O);                 // h2 -> d_out (final)
}
